// Decoder_27161373180016
// MI455X (gfx1250) — compile-verified, matrix operand order checked by reading, not run on hardware
//
#include <hip/hip_runtime.h>

// ---------------- types ----------------
typedef __bf16 bf16_t;
typedef __attribute__((ext_vector_type(16))) __bf16 bf16x16;
typedef __attribute__((ext_vector_type(8)))  float  f32x8;

union Frag { bf16x16 v; bf16_t h[16]; uint4 q[2]; };
union Acc  { f32x8 v; float f[8]; };

#define BB     32
#define NNODE  32
#define TT     20
#define DD     8
#define NEDGE  992          // 32*31
#define MSGD   64
#define EHIDD  256
#define OHIDD  256
#define NSTEP  19
#define BGRP   4            // batches co-processed per wave (B-fragment reuse)

__device__ inline float sigmoidf_(float x) { return 1.0f / (1.0f + __expf(-x)); }

__device__ inline void cvt8(Frag& f, int o, float4 a, float4 b) {
    f.h[o + 0] = (bf16_t)a.x; f.h[o + 1] = (bf16_t)a.y;
    f.h[o + 2] = (bf16_t)a.z; f.h[o + 3] = (bf16_t)a.w;
    f.h[o + 4] = (bf16_t)b.x; f.h[o + 5] = (bf16_t)b.y;
    f.h[o + 6] = (bf16_t)b.z; f.h[o + 7] = (bf16_t)b.w;
}

// packed weight fragment: 512 bf16 per fragment, lane-major 16 bf16 chunks
__device__ inline Frag loadWFrag(const bf16_t* __restrict__ base, int fragIdx, int lane) {
    Frag f;
    const bf16_t* p = base + (((size_t)fragIdx << 5) + lane) * 16;
    f.q[0] = *(const uint4*)p;
    f.q[1] = *(const uint4*)(p + 8);
    return f;
}

// A-fragment from a row-major bf16 row pointer in LDS (row = lane&15 chosen by caller)
__device__ inline Frag ldsAFrag(const bf16_t* rowPtr, int kb, int h) {
    Frag f;
    int k0 = kb * 32 + h * 8;
    f.q[0] = *(const uint4*)(rowPtr + k0);
    f.q[1] = *(const uint4*)(rowPtr + k0 + 16);
    return f;
}

__device__ inline void accZero(Acc& a) {
#pragma unroll
    for (int i = 0; i < 8; ++i) a.f[i] = 0.0f;
}

// ---------------- weight packing ----------------
// dst fragment layout: frag = nt*numKB + kb ; lane l: col = nt*16 + (l&15),
// element i: k_local = (l>>4)*8 + (i<8 ? i : 16 + (i-8)) — the 16-bit A/B
// VGPR striping from CDNA5 ISA.
__global__ void pack_frags(const float* __restrict__ W, bf16_t* __restrict__ dst,
                           int Kact, int Nact, int ld, int numKB) {
    int f = blockIdx.x;
    int nt = f / numKB, kb = f % numKB;
    int lane = threadIdx.x;
    int c = lane & 15, h = lane >> 4;
    bf16_t* out = dst + (((size_t)f << 5) + lane) * 16;
#pragma unroll
    for (int i = 0; i < 16; ++i) {
        int kl = (i < 8) ? (h * 8 + i) : (16 + h * 8 + (i - 8));
        int k = kb * 32 + kl;
        int n = nt * 16 + c;
        float v = (k < Kact && n < Nact) ? W[(size_t)k * ld + n] : 0.0f;
        out[i] = (bf16_t)v;
    }
}

__global__ void zero_f32(float* __restrict__ p, int n) {
    int i = blockIdx.x * blockDim.x + threadIdx.x;
    if (i < n) p[i] = 0.0f;
}

// ---------------- edge MLP (dominant GEMMs) ----------------
// One wave handles one 16-edge tile x BGRP batches: each packed-weight
// B fragment is loaded once and fed to BGRP WMMAs (4x less weight traffic).
#define EDGE_WAVES 2
__global__ __launch_bounds__(EDGE_WAVES * 32)
void edge_mlp_kernel(const float* __restrict__ hid,       // [B,32,64]
                     const float* __restrict__ edge_type, // [B,E,2]
                     const bf16_t* __restrict__ w1p,      // 2*16*4 frags
                     const bf16_t* __restrict__ w2p,      // 2*4*8 frags
                     const float* __restrict__ eb1,       // [2,256]
                     const float* __restrict__ eb2,       // [2,64]
                     float* __restrict__ agg)             // [B,32,64]
{
    extern __shared__ char smem[];
    const int lane = threadIdx.x & 31;
    const int w    = threadIdx.x >> 5;
    const int bg   = blockIdx.y;                    // batch group (4 batches)
    const int tile = blockIdx.x * EDGE_WAVES + w;   // 0..61 (exact)

    bf16_t* h1w = (bf16_t*)smem + (size_t)w * BGRP * 16 * EHIDD;  // 32KB/wave

    const int c = lane & 15;
    const int h = lane >> 4;

    // L2 warm-up for the shared weight fragments (global_prefetch_b8)
    __builtin_prefetch(w1p + (size_t)lane * 4096, 0, 0);
    __builtin_prefetch(w2p + (size_t)lane * 2048, 0, 0);

    // ---- edge structure for this tile (same for all batches)
    const int e = tile * 16 + c;                    // A-fragment row for lane
    int s = e / 31, q = e % 31;
    int r = q + (q >= s ? 1 : 0);

    int rrow[8];                                    // receivers of owned acc rows
#pragma unroll
    for (int i = 0; i < 8; ++i) {
        int m = (lane < 16) ? i : (i + 8);
        int ee = tile * 16 + m;
        int ss = ee / 31, qq = ee % 31;
        rrow[i] = qq + (qq >= ss ? 1 : 0);
    }

    // ---- resident A fragments: pre_msg = [sender | receiver] per batch
    Frag A[BGRP][4];
#pragma unroll
    for (int j = 0; j < BGRP; ++j) {
        const float* sh = hid + (((size_t)bg * BGRP + j) * NNODE + s) * MSGD;
        const float* rh = hid + (((size_t)bg * BGRP + j) * NNODE + r) * MSGD;
#pragma unroll
        for (int kb = 0; kb < 4; ++kb) {
#pragma unroll
            for (int part = 0; part < 2; ++part) {
                int k = kb * 32 + part * 16 + h * 8;
                const float* src = (k < MSGD) ? (sh + k) : (rh + (k - MSGD));
                float4 lo = *(const float4*)src;
                float4 hi = *(const float4*)(src + 4);
                cvt8(A[j][kb], part * 8, lo, hi);
            }
        }
    }

    for (int t = 0; t < 2; ++t) {
        // ---- GEMM1: [16,128]x4b @ [128,256] -> relu -> h1 (bf16 LDS)
        for (int nt = 0; nt < 16; ++nt) {
            Acc acc[BGRP];
#pragma unroll
            for (int j = 0; j < BGRP; ++j) accZero(acc[j]);
#pragma unroll
            for (int kb = 0; kb < 4; ++kb) {
                Frag Bf = loadWFrag(w1p, (t * 16 + nt) * 4 + kb, lane);
#pragma unroll
                for (int j = 0; j < BGRP; ++j)
                    acc[j].v = __builtin_amdgcn_wmma_f32_16x16x32_bf16(
                        false, A[j][kb].v, false, Bf.v, (short)0, acc[j].v, false, false);
            }
            float bias = eb1[t * EHIDD + nt * 16 + c];
#pragma unroll
            for (int j = 0; j < BGRP; ++j) {
#pragma unroll
                for (int i = 0; i < 8; ++i) {
                    float v = acc[j].f[i] + bias;
                    v = v > 0.0f ? v : 0.0f;
                    int row = (lane < 16) ? i : (i + 8);
                    h1w[j * 16 * EHIDD + row * EHIDD + nt * 16 + c] = (bf16_t)v;
                }
            }
        }

        // per-batch edge-type weights for this type (incl. /2 and /N_DIM)
        float wv[BGRP][8];
#pragma unroll
        for (int j = 0; j < BGRP; ++j) {
#pragma unroll
            for (int i = 0; i < 8; ++i) {
                int m = (lane < 16) ? i : (i + 8);
                int ee = tile * 16 + m;
                wv[j][i] = edge_type[(((size_t)bg * BGRP + j) * NEDGE + ee) * 2 + t]
                           * (1.0f / 16.0f);
            }
        }

        // ---- GEMM2: [16,256]x4b @ [256,64] -> relu -> weighted atomic scatter
        for (int nt2 = 0; nt2 < 4; ++nt2) {
            Acc acc[BGRP];
#pragma unroll
            for (int j = 0; j < BGRP; ++j) accZero(acc[j]);
#pragma unroll
            for (int kb2 = 0; kb2 < 8; ++kb2) {
                Frag Bf = loadWFrag(w2p, (t * 4 + nt2) * 8 + kb2, lane);
#pragma unroll
                for (int j = 0; j < BGRP; ++j) {
                    Frag Af = ldsAFrag(h1w + j * 16 * EHIDD + c * EHIDD, kb2, h);
                    acc[j].v = __builtin_amdgcn_wmma_f32_16x16x32_bf16(
                        false, Af.v, false, Bf.v, (short)0, acc[j].v, false, false);
                }
            }
            float bias = eb2[t * MSGD + nt2 * 16 + c];
#pragma unroll
            for (int j = 0; j < BGRP; ++j) {
#pragma unroll
                for (int i = 0; i < 8; ++i) {
                    float v = acc[j].f[i] + bias;
                    v = v > 0.0f ? v : 0.0f;
                    atomicAdd(&agg[(((size_t)bg * BGRP + j) * NNODE + rrow[i]) * MSGD
                                   + nt2 * 16 + c],
                              v * wv[j][i]);
                }
            }
        }
    }
}

// ---------------- GRU + output MLP ----------------
#define GRU_WAVES 2
__global__ __launch_bounds__(GRU_WAVES * 32)
void gru_out_kernel(const float* __restrict__ data,    // [B,32,20,8]
                    const float* __restrict__ agg,     // [B,32,64]
                    const float* __restrict__ hidOld,  // [B,32,64]
                    float* __restrict__ hidNew,        // [B,32,64]
                    const bf16_t* __restrict__ wip,    // 12 frags (K padded 8->32)
                    const bf16_t* __restrict__ whp,    // 24 frags
                    const bf16_t* __restrict__ o1p,    // 32 frags
                    const bf16_t* __restrict__ o2p,    // 128 frags
                    const bf16_t* __restrict__ owp,    // 8 frags (N padded 8->16)
                    const float* __restrict__ gbi, const float* __restrict__ gbh,
                    const float* __restrict__ ob1, const float* __restrict__ ob2,
                    const float* __restrict__ outb,
                    float* __restrict__ out, int step)
{
    extern __shared__ char smem[];
    const int lane = threadIdx.x & 31;
    const int w    = threadIdx.x >> 5;
    const int rt   = blockIdx.x * GRU_WAVES + w;   // 0..63 row tiles of 16

    const int c = lane & 15;
    const int h = lane >> 4;

    bf16_t* wbase = (bf16_t*)smem + (size_t)w * (16 * MSGD + 2 * 16 * OHIDD);
    bf16_t* hidL = wbase;                     // [16,64]
    bf16_t* x1L  = wbase + 16 * MSGD;         // [16,256]
    bf16_t* x2L  = x1L + 16 * OHIDD;          // [16,256]

    // ---- A fragments for agg (K=64) and ins (K=8 padded to 32)
    const int gr = rt * 16 + c;               // flattened (b*32+node) row
    const float* arow = agg + (size_t)gr * MSGD;
    Frag G[2];
#pragma unroll
    for (int kb = 0; kb < 2; ++kb)
#pragma unroll
        for (int part = 0; part < 2; ++part) {
            int k = kb * 32 + part * 16 + h * 8;
            float4 lo = *(const float4*)(arow + k);
            float4 hi = *(const float4*)(arow + k + 4);
            cvt8(G[kb], part * 8, lo, hi);
        }

    Frag insA;
#pragma unroll
    for (int i = 0; i < 16; ++i) insA.h[i] = (bf16_t)0.0f;
    if (h == 0) {  // k = 0..7 real, rest zero padding
        const float* ip = data + ((size_t)gr * TT + step) * DD;
        float4 lo = *(const float4*)ip;
        float4 hi = *(const float4*)(ip + 4);
        cvt8(insA, 0, lo, hi);
    }

    // ---- GRU gates per 16-column tile of MSG
    for (int jt = 0; jt < 4; ++jt) {
        Acc air, aiz, ain, ahr, ahz, ahn;
        accZero(air); accZero(aiz); accZero(ain);
        accZero(ahr); accZero(ahz); accZero(ahn);

        {   // gi = ins @ gWi (K padded to 32)
            Frag Br = loadWFrag(wip, jt,      lane);
            Frag Bz = loadWFrag(wip, jt + 4,  lane);
            Frag Bn = loadWFrag(wip, jt + 8,  lane);
            air.v = __builtin_amdgcn_wmma_f32_16x16x32_bf16(false, insA.v, false, Br.v, (short)0, air.v, false, false);
            aiz.v = __builtin_amdgcn_wmma_f32_16x16x32_bf16(false, insA.v, false, Bz.v, (short)0, aiz.v, false, false);
            ain.v = __builtin_amdgcn_wmma_f32_16x16x32_bf16(false, insA.v, false, Bn.v, (short)0, ain.v, false, false);
        }
#pragma unroll
        for (int kb = 0; kb < 2; ++kb) {  // gh = agg @ gWh
            Frag Br = loadWFrag(whp, (jt)     * 2 + kb, lane);
            Frag Bz = loadWFrag(whp, (jt + 4) * 2 + kb, lane);
            Frag Bn = loadWFrag(whp, (jt + 8) * 2 + kb, lane);
            ahr.v = __builtin_amdgcn_wmma_f32_16x16x32_bf16(false, G[kb].v, false, Br.v, (short)0, ahr.v, false, false);
            ahz.v = __builtin_amdgcn_wmma_f32_16x16x32_bf16(false, G[kb].v, false, Bz.v, (short)0, ahz.v, false, false);
            ahn.v = __builtin_amdgcn_wmma_f32_16x16x32_bf16(false, G[kb].v, false, Bn.v, (short)0, ahn.v, false, false);
        }
        int col = jt * 16 + c;
        float bir = gbi[col], biz = gbi[64 + col], bin = gbi[128 + col];
        float bhr = gbh[col], bhz = gbh[64 + col], bhn = gbh[128 + col];
#pragma unroll
        for (int i = 0; i < 8; ++i) {
            int m  = (lane < 16) ? i : (i + 8);
            int g2 = rt * 16 + m;
            float hold = hidOld[(size_t)g2 * MSGD + col];
            float rr = sigmoidf_(air.f[i] + bir + ahr.f[i] + bhr);
            float zz = sigmoidf_(aiz.f[i] + biz + ahz.f[i] + bhz);
            float nn = tanhf(ain.f[i] + bin + rr * (ahn.f[i] + bhn));
            float hn = (1.0f - zz) * nn + zz * hold;
            hidNew[(size_t)g2 * MSGD + col] = hn;
            hidL[m * MSGD + col] = (bf16_t)hn;
        }
    }

    // ---- x1 = relu(hidden_new @ oW1 + ob1)
    const bf16_t* hidRow = hidL + c * MSGD;
    for (int nt = 0; nt < 16; ++nt) {
        Acc acc; accZero(acc);
#pragma unroll
        for (int kb = 0; kb < 2; ++kb) {
            Frag Af = ldsAFrag(hidRow, kb, h);
            Frag Bf = loadWFrag(o1p, nt * 2 + kb, lane);
            acc.v = __builtin_amdgcn_wmma_f32_16x16x32_bf16(false, Af.v, false, Bf.v, (short)0, acc.v, false, false);
        }
        float bias = ob1[nt * 16 + c];
#pragma unroll
        for (int i = 0; i < 8; ++i) {
            float v = acc.f[i] + bias;
            v = v > 0.0f ? v : 0.0f;
            int row = (lane < 16) ? i : (i + 8);
            x1L[row * OHIDD + nt * 16 + c] = (bf16_t)v;
        }
    }

    // ---- x2 = relu(x1 @ oW2 + ob2)
    const bf16_t* x1Row = x1L + c * OHIDD;
    for (int nt = 0; nt < 16; ++nt) {
        Acc acc; accZero(acc);
#pragma unroll
        for (int kb = 0; kb < 8; ++kb) {
            Frag Af = ldsAFrag(x1Row, kb, h);
            Frag Bf = loadWFrag(o2p, nt * 8 + kb, lane);
            acc.v = __builtin_amdgcn_wmma_f32_16x16x32_bf16(false, Af.v, false, Bf.v, (short)0, acc.v, false, false);
        }
        float bias = ob2[nt * 16 + c];
#pragma unroll
        for (int i = 0; i < 8; ++i) {
            float v = acc.f[i] + bias;
            v = v > 0.0f ? v : 0.0f;
            int row = (lane < 16) ? i : (i + 8);
            x2L[row * OHIDD + nt * 16 + c] = (bf16_t)v;
        }
    }

    // ---- pred = ins + x2 @ outW + outb   (N padded 8 -> 16)
    {
        const bf16_t* x2Row = x2L + c * OHIDD;
        Acc acc; accZero(acc);
#pragma unroll
        for (int kb = 0; kb < 8; ++kb) {
            Frag Af = ldsAFrag(x2Row, kb, h);
            Frag Bf = loadWFrag(owp, kb, lane);
            acc.v = __builtin_amdgcn_wmma_f32_16x16x32_bf16(false, Af.v, false, Bf.v, (short)0, acc.v, false, false);
        }
        if (c < DD) {
            float bias = outb[c];
#pragma unroll
            for (int i = 0; i < 8; ++i) {
                int m  = (lane < 16) ? i : (i + 8);
                int g2 = rt * 16 + m;
                float ins = data[((size_t)g2 * TT + step) * DD + c];
                out[((size_t)g2 * NSTEP + step) * DD + c] = ins + acc.f[i] + bias;
            }
        }
    }
}

// ---------------- host driver ----------------
extern "C" void kernel_launch(void* const* d_in, const int* in_sizes, int n_in,
                              void* d_out, int out_size, void* d_ws, size_t ws_size,
                              hipStream_t stream) {
    (void)in_sizes; (void)n_in; (void)out_size; (void)ws_size;
    const float* data      = (const float*)d_in[0];
    const float* edge_type = (const float*)d_in[1];
    // d_in[2] rec_rel / d_in[3] send_rel: fully-connected off-diagonal, closed form
    const float* eW1  = (const float*)d_in[4];
    const float* eb1  = (const float*)d_in[5];
    const float* eW2  = (const float*)d_in[6];
    const float* eb2  = (const float*)d_in[7];
    const float* gWi  = (const float*)d_in[8];
    const float* gbi  = (const float*)d_in[9];
    const float* gWh  = (const float*)d_in[10];
    const float* gbh  = (const float*)d_in[11];
    const float* oW1  = (const float*)d_in[12];
    const float* ob1  = (const float*)d_in[13];
    const float* oW2  = (const float*)d_in[14];
    const float* ob2  = (const float*)d_in[15];
    const float* outW = (const float*)d_in[16];
    const float* outb = (const float*)d_in[17];
    float* out = (float*)d_out;

    // workspace carve-up (bf16 packed weights, then f32 state) ~1.2 MB total
    bf16_t* wp = (bf16_t*)d_ws;
    size_t off = 0;
    bf16_t* w1p = wp + off; off += (size_t)2 * 16 * 4 * 512;
    bf16_t* w2p = wp + off; off += (size_t)2 * 4 * 8 * 512;
    bf16_t* whp = wp + off; off += (size_t)12 * 2 * 512;
    bf16_t* wip = wp + off; off += (size_t)12 * 1 * 512;
    bf16_t* o1p = wp + off; off += (size_t)16 * 2 * 512;
    bf16_t* o2p = wp + off; off += (size_t)16 * 8 * 512;
    bf16_t* owp = wp + off; off += (size_t)1 * 8 * 512;
    size_t foff = (off * sizeof(bf16_t) + 255) & ~(size_t)255;
    float* hidA = (float*)((char*)d_ws + foff);
    float* hidB = hidA + (size_t)BB * NNODE * MSGD;
    float* agg  = hidB + (size_t)BB * NNODE * MSGD;
    const int HN = BB * NNODE * MSGD;  // 65536

    // pack all weights into WMMA fragment layout (bf16)
    for (int t = 0; t < 2; ++t) {
        pack_frags<<<16 * 4, 32, 0, stream>>>(eW1 + (size_t)t * 128 * 256,
                                              w1p + (size_t)t * 16 * 4 * 512, 128, 256, 256, 4);
        pack_frags<<<4 * 8, 32, 0, stream>>>(eW2 + (size_t)t * 256 * 64,
                                             w2p + (size_t)t * 4 * 8 * 512, 256, 64, 64, 8);
    }
    pack_frags<<<12 * 2, 32, 0, stream>>>(gWh, whp, 64, 192, 192, 2);
    pack_frags<<<12 * 1, 32, 0, stream>>>(gWi, wip, 8, 192, 192, 1);   // K pad 8->32
    pack_frags<<<16 * 2, 32, 0, stream>>>(oW1, o1p, 64, 256, 256, 2);
    pack_frags<<<16 * 8, 32, 0, stream>>>(oW2, o2p, 256, 256, 256, 8);
    pack_frags<<<1 * 8, 32, 0, stream>>>(outW, owp, 256, 8, 8, 8);     // N pad 8->16
    zero_f32<<<(HN + 255) / 256, 256, 0, stream>>>(hidA, HN);

    float* hin = hidA;
    float* hout = hidB;
    const size_t edgeLds = (size_t)EDGE_WAVES * BGRP * 16 * EHIDD * sizeof(bf16_t);           // 64 KB
    const size_t gruLds  = (size_t)GRU_WAVES * (16 * MSGD + 2 * 16 * OHIDD) * sizeof(bf16_t); // 36 KB

    for (int step = 0; step < NSTEP; ++step) {
        zero_f32<<<(HN + 255) / 256, 256, 0, stream>>>(agg, HN);
        dim3 eg((NEDGE / 16) / EDGE_WAVES, BB / BGRP);  // (31, 8)
        edge_mlp_kernel<<<eg, EDGE_WAVES * 32, edgeLds, stream>>>(
            hin, edge_type, w1p, w2p, eb1, eb2, agg);
        gru_out_kernel<<<(BB * NNODE / 16) / GRU_WAVES, GRU_WAVES * 32, gruLds, stream>>>(
            data, agg, hin, hout, wip, whp, o1p, o2p, owp,
            gbi, gbh, ob1, ob2, outb, out, step);
        float* tmp = hin; hin = hout; hout = tmp;
    }
}
